// CrossAttention_44049184587924
// MI455X (gfx1250) — matrix operand order repaired and checked by reading, not compiled, not run on hardware
//
#include <hip/hip_runtime.h>

typedef __attribute__((ext_vector_type(16))) __bf16 v16bf;
typedef __attribute__((ext_vector_type(8)))  __bf16 v8bf;
typedef __attribute__((ext_vector_type(4)))  __bf16 v4bf;
typedef __attribute__((ext_vector_type(8)))  float  v8f;
typedef __attribute__((ext_vector_type(4)))  float  v4f;

#define DEVINL __device__ __forceinline__

static constexpr int Bn = 2, LQc = 1024, LKc = 10000, Ec = 256, Hc = 8, Dc = 32;
static constexpr int QT = LQc / 16;          // 64 q-tiles per (b,h)
static constexpr int NT = (LKc + 63) / 64;   // 157 key-tiles of 64
static constexpr int NSPLIT = 8;             // split-K factor
static constexpr int TPP = (NT + NSPLIT - 1) / NSPLIT;  // 20 tiles / partition
static constexpr int NPART = Bn * Hc * QT * NSPLIT;     // 8192 partials
static constexpr float NEG_BIG = -3.0e38f;
static constexpr float LOG2E = 1.44269504088896340736f;

DEVINL v8f wmma_bf16(v16bf a, v16bf b, v8f c) {
  return __builtin_amdgcn_wmma_f32_16x16x32_bf16(false, a, false, b, (short)0, c,
                                                 false, false);
}

DEVINL v16bf load_frag16(const __bf16* p0, const __bf16* p1) {
  v8bf a = *(const v8bf*)p0;
  v8bf b = *(const v8bf*)p1;
  v16bf f;
#pragma unroll
  for (int i = 0; i < 8; ++i) { f[i] = a[i]; f[8 + i] = b[i]; }
  return f;
}

DEVINL float fexp2(float x) { return __builtin_amdgcn_exp2f(x); }

// Fused single-instruction DPP16 max-combine steps: dst = max(dpp(v), v).
DEVINL float dppmax_x1(float v) {
  float r;
  asm("v_max_num_f32_dpp %0, %1, %1 quad_perm:[1,0,3,2] row_mask:0xf bank_mask:0xf bound_ctrl:1"
      : "=v"(r) : "v"(v));
  return r;
}
DEVINL float dppmax_x2(float v) {
  float r;
  asm("v_max_num_f32_dpp %0, %1, %1 quad_perm:[2,3,0,1] row_mask:0xf bank_mask:0xf bound_ctrl:1"
      : "=v"(r) : "v"(v));
  return r;
}
DEVINL float dppmax_hm(float v) {
  float r;
  asm("v_max_num_f32_dpp %0, %1, %1 row_half_mirror row_mask:0xf bank_mask:0xf bound_ctrl:1"
      : "=v"(r) : "v"(v));
  return r;
}
DEVINL float dppmax_m(float v) {
  float r;
  asm("v_max_num_f32_dpp %0, %1, %1 row_mirror row_mask:0xf bank_mask:0xf bound_ctrl:1"
      : "=v"(r) : "v"(v));
  return r;
}

// ---------------------------------------------------------------------------
// Elementwise fp32 -> bf16 (coalesced, 4 elems/thread).
// ---------------------------------------------------------------------------
__global__ void cvt_bf16_kernel(const float* __restrict__ in,
                                __bf16* __restrict__ out, int n4) {
  int id = blockIdx.x * blockDim.x + threadIdx.x;
  if (id >= n4) return;
  v4f x = *(const v4f*)(in + (size_t)id * 4);
  v4bf y;
#pragma unroll
  for (int i = 0; i < 4; ++i) y[i] = (__bf16)x[i];
  *(v4bf*)(out + (size_t)id * 4) = y;
}

// ---------------------------------------------------------------------------
// Weight prep: W fp32 [256(k),256(n)] -> Wt bf16 [256(n),256(k)].
// ---------------------------------------------------------------------------
__global__ void transpose_w_kernel(const float* __restrict__ W,
                                   __bf16* __restrict__ Wt) {
  int id = blockIdx.x * blockDim.x + threadIdx.x;  // 65536 threads
  int k = id >> 8, n = id & 255;
  Wt[(size_t)n * Ec + k] = (__bf16)W[(size_t)k * Ec + n];
}

// ---------------------------------------------------------------------------
// Projection: out[M,256] = (X_bf16[M,256] @ W + bias) * oscale.
// W given as Wt bf16 [n][k].  oscale = log2(e) for Q (exp2-domain softmax).
// MODE 0: bf16 row-major out   MODE 1: bf16 transposed out (V)
// MODE 2: f32 row-major out (final). One wave = one 16x16 tile, 8 WMMAs.
// ---------------------------------------------------------------------------
template <int MODE>
__global__ void proj_gemm_kernel(const __bf16* __restrict__ X,
                                 const __bf16* __restrict__ Wt,
                                 const float* __restrict__ bias,
                                 void* __restrict__ out,
                                 int M, int rowsPerBatch, float oscale) {
  const int wave = (int)((blockIdx.x * blockDim.x + threadIdx.x) >> 5);
  const int lane = (int)(threadIdx.x & 31);
  const int mt = wave >> 4;
  const int nt = wave & 15;
  if (mt * 16 >= M) return;          // wave-uniform
  const int m0 = mt * 16, n0 = nt * 16;
  const int hi = lane >> 4, l15 = lane & 15;
  const int c0 = hi * 8;

  v8f acc = {0.f, 0.f, 0.f, 0.f, 0.f, 0.f, 0.f, 0.f};
  const __bf16* xrow = X + (size_t)(m0 + l15) * Ec;
  const __bf16* wrow = Wt + (size_t)(n0 + l15) * Ec;

#pragma unroll
  for (int kt = 0; kt < Ec; kt += 32) {
    v16bf a = load_frag16(xrow + kt + c0, xrow + kt + c0 + 16);
    v16bf b = load_frag16(wrow + kt + c0, wrow + kt + c0 + 16);
    acc = wmma_bf16(a, b, acc);
  }

  const float bv = bias[n0 + l15];
#pragma unroll
  for (int r = 0; r < 8; ++r) {
    const int row = m0 + r + 8 * hi;
    const int col = n0 + l15;
    const float v = (acc[r] + bv) * oscale;
    if (MODE == 0) {
      ((__bf16*)out)[(size_t)row * Ec + col] = (__bf16)v;
    } else if (MODE == 1) {
      const int bb = row / rowsPerBatch;
      const int kp = row - bb * rowsPerBatch;
      ((__bf16*)out)[((size_t)bb * Ec + col) * rowsPerBatch + kp] = (__bf16)v;
    } else {
      ((float*)out)[(size_t)row * Ec + col] = v;
    }
  }
}

// ---------------------------------------------------------------------------
// Shared tile epilogue (exp2 domain): online softmax with fused DPP max
// reduction, P bounce through LDS, V fragment loads, and 6 WMMAs
// (2 row-sum P x ones + 4 accum P x V).
// ---------------------------------------------------------------------------
DEVINL void fa_accumulate(v8f s[4],
                          const __bf16* vA0p, const __bf16* vA1p,
                          const __bf16* vB0p, const __bf16* vB1p,
                          __bf16* myP, int hi, int l15, int c0, v16bf ones,
                          float (&mstat)[8], v8f& acc0, v8f& acc1, v8f& lacc) {
  // Row maxes: local 4-way max, then 4 fused DPP16 steps, batched across rows.
  float vmax[8];
#pragma unroll
  for (int r = 0; r < 8; ++r)
    vmax[r] = fmaxf(fmaxf(s[0][r], s[1][r]), fmaxf(s[2][r], s[3][r]));
#pragma unroll
  for (int r = 0; r < 8; ++r) vmax[r] = dppmax_x1(vmax[r]);
#pragma unroll
  for (int r = 0; r < 8; ++r) vmax[r] = dppmax_x2(vmax[r]);
#pragma unroll
  for (int r = 0; r < 8; ++r) vmax[r] = dppmax_hm(vmax[r]);
#pragma unroll
  for (int r = 0; r < 8; ++r) vmax[r] = dppmax_m(vmax[r]);

#pragma unroll
  for (int r = 0; r < 8; ++r) {
    const float mn = fmaxf(mstat[r], vmax[r]);
    const float sc = fexp2(mstat[r] - mn);   // scores are in log2 units
    mstat[r] = mn;
    acc0[r] *= sc; acc1[r] *= sc; lacc[r] *= sc;
#pragma unroll
    for (int j = 0; j < 4; ++j) s[j][r] = fexp2(s[j][r] - mn);
  }

  // P: C-layout -> A-layout via LDS (16x64 bf16)
#pragma unroll
  for (int r = 0; r < 8; ++r) {
    const int rowoff = (r + 8 * hi) * 64;
#pragma unroll
    for (int j = 0; j < 4; ++j)
      myP[rowoff + 16 * j + l15] = (__bf16)s[j][r];
  }
  const __bf16* prr = myP + l15 * 64;
  v16bf pfA = load_frag16(prr + 16 * hi, prr + 16 * hi + 8);            // k = 0..31
  v16bf pfB = load_frag16(prr + 32 + 16 * hi, prr + 32 + 16 * hi + 8);  // k = 32..63

  v16bf vA0 = load_frag16(vA0p, vA0p + 8);
  v16bf vA1 = load_frag16(vA1p, vA1p + 8);
  v16bf vB0 = load_frag16(vB0p, vB0p + 8);
  v16bf vB1 = load_frag16(vB1p, vB1p + 8);

  lacc = wmma_bf16(pfA, ones, lacc);   // row-sums on the matrix pipe
  lacc = wmma_bf16(pfB, ones, lacc);
  acc0 = wmma_bf16(pfA, vA0, acc0);
  acc1 = wmma_bf16(pfA, vA1, acc1);
  acc0 = wmma_bf16(pfB, vB0, acc0);
  acc1 = wmma_bf16(pfB, vB1, acc1);
}

// ---------------------------------------------------------------------------
// Split-K flash attention. Wave = one (b,h,qt,partition).
// Full tiles use incrementing pointers + immediate offsets (no clamps);
// only the globally-last tile runs a masked, clamped iteration.
// ---------------------------------------------------------------------------
__global__ void flash_attn_part_kernel(const __bf16* __restrict__ Qb,
                                       const __bf16* __restrict__ Kb,
                                       const __bf16* __restrict__ Vt,
                                       float* __restrict__ Opart,
                                       float* __restrict__ Mpart,
                                       float* __restrict__ Lpart) {
  __shared__ __align__(16) __bf16 sP[4][16 * 64];  // 2KB / wave
  const int wib  = (int)(threadIdx.x >> 5);
  const int lane = (int)(threadIdx.x & 31);
  const int wave = (int)(blockIdx.x * (blockDim.x >> 5) + wib);
  if (wave >= NPART) return;
  const int p  = wave % NSPLIT;
  int rest     = wave / NSPLIT;
  const int qt = rest % QT;
  rest /= QT;
  const int h  = rest % Hc;
  const int b  = rest / Hc;

  const int hi = lane >> 4, l15 = lane & 15;
  const int m0 = qt * 16;
  const int d0 = h * Dc;
  const int c0 = hi * 8;
  __bf16* myP = sP[wib];

  const int t0 = p * TPP;
  const int t1 = (t0 + TPP < NT) ? t0 + TPP : NT;
  const int t1f = (t1 == NT) ? NT - 1 : t1;      // full tiles only

  const __bf16* Kbase  = Kb + (size_t)b * LKc * Ec + d0 + 16 * hi;
  const __bf16* V0base = Vt + ((size_t)b * Ec + d0 + l15) * LKc;
  const __bf16* V1base = Vt + ((size_t)b * Ec + d0 + 16 + l15) * LKc;

  const __bf16* qrow = Qb + (size_t)(b * LQc + m0 + l15) * Ec + d0;
  v16bf qf = load_frag16(qrow + 16 * hi, qrow + 16 * hi + 8);

  v16bf ones;
#pragma unroll
  for (int i = 0; i < 16; ++i) ones[i] = (__bf16)1.0f;

  float mstat[8];
  v8f acc0 = {0.f, 0.f, 0.f, 0.f, 0.f, 0.f, 0.f, 0.f};
  v8f acc1 = acc0;
  v8f lacc = acc0;
#pragma unroll
  for (int r = 0; r < 8; ++r) mstat[r] = NEG_BIG;

  // Incrementing pointers; all fragment loads become immediate offsets.
  const __bf16* kptr = Kbase + ((size_t)t0 * 64 + l15) * Ec;
  const __bf16* v0p  = V0base + t0 * 64 + 16 * hi;
  const __bf16* v1p  = V1base + t0 * 64 + 16 * hi;

  for (int kt = t0; kt < t1f; ++kt) {
    // Prefetch next tile's K rows (immediate-offset global_prefetch_b8).
    __builtin_prefetch(kptr + (size_t)64 * Ec, 0, 1);
    __builtin_prefetch(kptr + (size_t)96 * Ec, 0, 1);

    v8f s[4];
#pragma unroll
    for (int j = 0; j < 4; ++j) {
      v16bf kf = load_frag16(kptr + (size_t)j * 16 * Ec,
                             kptr + (size_t)j * 16 * Ec + 8);
      v8f z = {0.f, 0.f, 0.f, 0.f, 0.f, 0.f, 0.f, 0.f};
      s[j] = wmma_bf16(qf, kf, z);
    }

    fa_accumulate(s, v0p, v1p, v0p + 32, v1p + 32,
                  myP, hi, l15, c0, ones, mstat, acc0, acc1, lacc);

    kptr += (size_t)64 * Ec;
    v0p  += 64;
    v1p  += 64;
  }

  // Masked tail tile (only the partition owning tile NT-1 reaches here).
  if (t1 == NT) {
    const int kp0 = (NT - 1) * 64;
    v8f s[4];
#pragma unroll
    for (int j = 0; j < 4; ++j) {
      int r0 = kp0 + 16 * j + l15; if (r0 >= LKc) r0 = LKc - 1;
      const __bf16* kp = Kbase + (size_t)r0 * Ec;
      v16bf kf = load_frag16(kp, kp + 8);
      v8f z = {0.f, 0.f, 0.f, 0.f, 0.f, 0.f, 0.f, 0.f};
      s[j] = wmma_bf16(qf, kf, z);
    }
#pragma unroll
    for (int j = 0; j < 4; ++j) {
      const bool ok = (kp0 + 16 * j + l15) < LKc;
      if (!ok)
#pragma unroll
        for (int r = 0; r < 8; ++r) s[j][r] = NEG_BIG;
    }
    int cbA = kp0 + 16 * hi;      if (cbA > LKc - 16) cbA = LKc - 16;
    int cbB = kp0 + 32 + 16 * hi; if (cbB > LKc - 16) cbB = LKc - 16;
    fa_accumulate(s, V0base + cbA, V1base + cbA, V0base + cbB, V1base + cbB,
                  myP, hi, l15, c0, ones, mstat, acc0, acc1, lacc);
  }

  // ---- emit partial: O fp32 [16][32], m[16], l[16] ----
  float* op = Opart + (size_t)wave * 512;
#pragma unroll
  for (int r = 0; r < 8; ++r) {
    const int row = r + 8 * hi;
    op[row * 32 + l15]      = acc0[r];
    op[row * 32 + 16 + l15] = acc1[r];
    if (l15 == r) {
      Mpart[(size_t)wave * 16 + row] = mstat[r];
      Lpart[(size_t)wave * 16 + row] = lacc[r];
    }
  }
}

// ---------------------------------------------------------------------------
// Merge: one wave per (b,h,qt). Lane = d-column, loop over 16 rows,
// log-sum-exp (base-2) combine of NSPLIT partials; write bf16 attn output.
// ---------------------------------------------------------------------------
__global__ void flash_merge_kernel(const float* __restrict__ Opart,
                                   const float* __restrict__ Mpart,
                                   const float* __restrict__ Lpart,
                                   __bf16* __restrict__ Ob) {
  const int lane = (int)(threadIdx.x & 31);
  const int wave = (int)((blockIdx.x * blockDim.x + threadIdx.x) >> 5);
  if (wave >= Bn * Hc * QT) return;
  const int qt = wave % QT;
  int rest = wave / QT;
  const int h = rest % Hc;
  const int b = rest / Hc;
  const size_t base = (size_t)wave * NSPLIT;

  for (int row = 0; row < 16; ++row) {
    float mg = NEG_BIG;
#pragma unroll
    for (int pp = 0; pp < NSPLIT; ++pp)
      mg = fmaxf(mg, Mpart[(base + pp) * 16 + row]);
    float osum = 0.f, lsum = 0.f;
#pragma unroll
    for (int pp = 0; pp < NSPLIT; ++pp) {
      const float w = fexp2(Mpart[(base + pp) * 16 + row] - mg);  // log2 units
      lsum += w * Lpart[(base + pp) * 16 + row];
      osum += w * Opart[(base + pp) * 512 + row * 32 + lane];
    }
    const float inv = __builtin_amdgcn_rcpf(lsum);
    Ob[(size_t)(b * LQc + qt * 16 + row) * Ec + h * Dc + lane] =
        (__bf16)(osum * inv);
  }
}

// ---------------------------------------------------------------------------
extern "C" void kernel_launch(void* const* d_in, const int* in_sizes, int n_in,
                              void* d_out, int out_size, void* d_ws, size_t ws_size,
                              hipStream_t stream) {
  const float* bev     = (const float*)d_in[0];
  const float* queries = (const float*)d_in[1];
  const float* Wq = (const float*)d_in[2];
  const float* bq = (const float*)d_in[3];
  const float* Wk = (const float*)d_in[4];
  const float* bk = (const float*)d_in[5];
  const float* Wv = (const float*)d_in[6];
  const float* bv = (const float*)d_in[7];
  const float* Wo = (const float*)d_in[8];
  const float* bo = (const float*)d_in[9];

  char* ws = (char*)d_ws;
  const size_t bev_bytes = (size_t)Bn * LKc * Ec * 2;   // 10.24 MB
  const size_t qin_bytes = (size_t)Bn * LQc * Ec * 2;   // 1.0 MB
  const size_t w_bytes   = (size_t)Ec * Ec * 2;         // 128 KB each

  size_t off = 0;
  __bf16* bevB = (__bf16*)(ws + off); off += bev_bytes;
  __bf16* qinB = (__bf16*)(ws + off); off += qin_bytes;
  __bf16* WqT  = (__bf16*)(ws + off); off += w_bytes;
  __bf16* WkT  = (__bf16*)(ws + off); off += w_bytes;
  __bf16* WvT  = (__bf16*)(ws + off); off += w_bytes;
  __bf16* WoT  = (__bf16*)(ws + off); off += w_bytes;
  __bf16* Qb   = (__bf16*)(ws + off); off += qin_bytes;
  __bf16* Kb   = (__bf16*)(ws + off); off += bev_bytes;
  __bf16* Vtb  = (__bf16*)(ws + off); off += bev_bytes;
  __bf16* attO = (__bf16*)(ws + off); off += qin_bytes;
  float* Opart = (float*)(ws + off);  off += (size_t)NPART * 512 * 4;  // 16.8 MB
  float* Mpart = (float*)(ws + off);  off += (size_t)NPART * 16 * 4;
  float* Lpart = (float*)(ws + off);  off += (size_t)NPART * 16 * 4;

  const int Mq  = Bn * LQc;   // 2048
  const int Mkv = Bn * LKc;   // 20000

  // 1) one-time conversions / weight prep
  {
    int n4b = (Bn * LKc * Ec) / 4;
    int n4q = (Bn * LQc * Ec) / 4;
    cvt_bf16_kernel<<<(n4b + 255) / 256, 256, 0, stream>>>(bev, bevB, n4b);
    cvt_bf16_kernel<<<(n4q + 255) / 256, 256, 0, stream>>>(queries, qinB, n4q);
    transpose_w_kernel<<<256, 256, 0, stream>>>(Wq, WqT);
    transpose_w_kernel<<<256, 256, 0, stream>>>(Wk, WkT);
    transpose_w_kernel<<<256, 256, 0, stream>>>(Wv, WvT);
    transpose_w_kernel<<<256, 256, 0, stream>>>(Wo, WoT);
  }

  // 2) QKV projections (Q pre-scaled by log2(e) for exp2-domain softmax)
  {
    dim3 blk(128);
    dim3 gq((Mq + 3) / 4);
    dim3 gk((Mkv + 3) / 4);
    proj_gemm_kernel<0><<<gq, blk, 0, stream>>>(qinB, WqT, bq, (void*)Qb,  Mq,  0,  LOG2E);
    proj_gemm_kernel<0><<<gk, blk, 0, stream>>>(bevB, WkT, bk, (void*)Kb,  Mkv, 0,  1.0f);
    proj_gemm_kernel<1><<<gk, blk, 0, stream>>>(bevB, WvT, bv, (void*)Vtb, Mkv, LKc, 1.0f);
  }

  // 3) split-K flash attention + merge
  {
    dim3 blk(128);
    dim3 gpart((NPART + 3) / 4);
    flash_attn_part_kernel<<<gpart, blk, 0, stream>>>(Qb, Kb, Vtb,
                                                      Opart, Mpart, Lpart);
    dim3 gmerge((Bn * Hc * QT + 3) / 4);
    flash_merge_kernel<<<gmerge, blk, 0, stream>>>(Opart, Mpart, Lpart, attO);
  }

  // 4) output projection -> fp32 d_out
  {
    dim3 blk(128);
    dim3 go((Mq + 3) / 4);
    proj_gemm_kernel<2><<<go, blk, 0, stream>>>(attO, WoT, bo, d_out, Mq, 0, 1.0f);
  }
}
